// ADMAttentionBlock_66425964200225
// MI455X (gfx1250) — matrix operand order repaired and checked by reading, not compiled, not run on hardware
//
#include <hip/hip_runtime.h>
#include <math.h>

// ---------------------------------------------------------------------------
// ADM attention block for MI455X (gfx1250, wave32, WMMA 16x16x32 bf16).
// B=4, C=512, N=4096. All large GEMMs (qkv, QK^T, PV, out-proj) run on
// v_wmma_f32_16x16x32_bf16 with f32 accumulation. Weight / V / hT tile
// staging uses the Tensor Data Mover (tensor_load_to_lds + s_wait_tensorcnt)
// where the copy is a pure 2D tile move; transposing stages stay manual.
// ---------------------------------------------------------------------------

typedef __attribute__((ext_vector_type(16))) __bf16         v16bf;
typedef __attribute__((ext_vector_type(8)))  float          v8f;
typedef __attribute__((ext_vector_type(4)))  unsigned int   u32x4;
typedef __attribute__((ext_vector_type(8)))  int            i32x8;
typedef __attribute__((ext_vector_type(4)))  int            i32x4;
typedef __attribute__((ext_vector_type(4)))  float          f32x4;

struct FragBits { u32x4 lo, hi; };

#if __has_builtin(__builtin_amdgcn_tensor_load_to_lds) && __has_builtin(__builtin_amdgcn_s_wait_tensorcnt)
#define HAVE_TDM 1
#else
#define HAVE_TDM 0
#endif

__device__ __forceinline__ unsigned short f2bf(float f) {
  unsigned int u = __builtin_bit_cast(unsigned int, f);
  u += 0x7FFFu + ((u >> 16) & 1u);   // round-to-nearest-even
  return (unsigned short)(u >> 16);
}

#if HAVE_TDM
// 2D tensor tile -> LDS via TDM (CDNA5 ISA).
// Elements are 2 bytes (data_size=1). Dims/strides in elements.
// pad_interval/pad_amount: encoded per D# group1 (interval: 0=2 DW .. 7=256 DW;
// amount: 0=1 DW .. 127=128 DW). Must be issued wave-uniform (one wave).
__device__ __forceinline__ void tdm_load_2d(unsigned lds_byte_off, const void* gaddr,
                                            unsigned tensor_d0, unsigned tensor_d1,
                                            unsigned tile_d0, unsigned tile_d1,
                                            unsigned long long stride0,
                                            unsigned pad_interval, unsigned pad_amount,
                                            unsigned pad_en) {
  unsigned long long ga = (unsigned long long)(size_t)gaddr;
  u32x4 g0;
  g0[0] = 1u;                                   // count=1 (valid), no gather
  g0[1] = lds_byte_off;                         // LDS destination (bytes)
  g0[2] = (unsigned)ga;                         // global_addr[31:0]
  g0[3] = (unsigned)(ga >> 32) | (2u << 30);    // global_addr[56:32] | type=2
  i32x8 g1;
  g1[0] = (int)((1u << 16) | (pad_en << 20) | (pad_interval << 22) | (pad_amount << 25));
  g1[1] = (int)(tensor_d0 << 16);                              // dim0[15:0] @63:48
  g1[2] = (int)((tensor_d0 >> 16) | (tensor_d1 << 16));        // dim0[31:16], dim1[15:0]
  g1[3] = (int)((tensor_d1 >> 16) | (tile_d0 << 16));          // dim1[31:16], tile0
  g1[4] = (int)tile_d1;                                        // tile1, tile2=0
  g1[5] = (int)(unsigned)stride0;                              // stride0[31:0]
  g1[6] = (int)(unsigned)(stride0 >> 32);                      // stride0[47:32], stride1=0
  g1[7] = 0;
  i32x4 z4 = {0, 0, 0, 0};
#if __clang_major__ >= 23
  i32x8 z8 = {0, 0, 0, 0, 0, 0, 0, 0};
  __builtin_amdgcn_tensor_load_to_lds(g0, g1, z4, z4, z8, 0);
#else
  __builtin_amdgcn_tensor_load_to_lds(g0, g1, z4, z4, 0);
#endif
}

__device__ __forceinline__ unsigned lds_off(const void* p) {
  // Generic pointers to LDS carry the byte offset in addr[31:0] (ISA §10.2).
  return (unsigned)(size_t)p;
}
#endif

// A fragment: A[m][k] (16x32 bf16), LDS layout [m][k], k contiguous.
// Per ISA: lane<16 holds K 0-7 & 16-23; lane>=16 holds K 8-15 & 24-31.
__device__ __forceinline__ v16bf load_frag_a(const unsigned short* base, int ld,
                                             int row0, int lane) {
  int r  = row0 + (lane & 15);
  int kb = (lane & 16) ? 8 : 0;
  const unsigned short* p = base + r * ld + kb;
  FragBits f;
  f.lo = *(const u32x4*)(p);
  f.hi = *(const u32x4*)(p + 16);
  return __builtin_bit_cast(v16bf, f);
}

// B fragment: B[k][n] (32x16 bf16), LDS layout [n][k], k contiguous.
// Per ISA: lane<16 holds K 0-7 & 16-23; lane>=16 holds K 8-15 & 24-31.
__device__ __forceinline__ v16bf load_frag_b(const unsigned short* base, int ld,
                                             int col0, int lane) {
  int c  = col0 + (lane & 15);
  int kb = (lane & 16) ? 8 : 0;
  const unsigned short* p = base + c * ld + kb;
  FragBits f;
  f.lo = *(const u32x4*)(p);
  f.hi = *(const u32x4*)(p + 16);
  return __builtin_bit_cast(v16bf, f);
}

#define WMMA_BF16(a, b, c) \
  __builtin_amdgcn_wmma_f32_16x16x32_bf16(false, (a), false, (b), (short)0, (c), false, false)

// ---------------------------------------------------------------------------
// Kernel 1: weight norm (dim=0): w[r] = g[r] * v[r] / ||v[r]||, emit bf16.
// ---------------------------------------------------------------------------
__global__ __launch_bounds__(128, 1)
void adm_wnorm(const float* __restrict__ v, const float* __restrict__ g,
               unsigned short* __restrict__ w, int cols) {
  __shared__ float red[128];
  int r = blockIdx.x;
  int t = threadIdx.x;
  const float* row = v + (size_t)r * cols;
  float s = 0.f;
  for (int c = t; c < cols; c += 128) { float x = row[c]; s += x * x; }
  red[t] = s;
  __syncthreads();
  for (int off = 64; off > 0; off >>= 1) {
    if (t < off) red[t] += red[t + off];
    __syncthreads();
  }
  float scale = g[r] * rsqrtf(red[0]);
  unsigned short* wr = w + (size_t)r * cols;
  for (int c = t; c < cols; c += 128) wr[c] = f2bf(row[c] * scale);
}

// ---------------------------------------------------------------------------
// Kernel 2: qkv = w_qkv[1536x512] @ x[b][512][4096] -> f32 [b][1536][4096]
// 64x64 block tile, 4 waves, 2x2 WMMA tiles per wave, K step 32.
// A tile (64x512 bf16 weights) staged ONCE via TDM into padded LDS (ld 520).
// ---------------------------------------------------------------------------
__global__ __launch_bounds__(128, 1)
void adm_qkv_gemm(const unsigned short* __restrict__ wq,
                  const float* __restrict__ x,
                  float* __restrict__ qkv) {
  __shared__ unsigned short sA[64 * 520];  // [m][k] full K extent
  __shared__ unsigned short sB[64 * 40];   // [n][k] per K-step
  const int b  = blockIdx.z;
  const int m0 = blockIdx.y * 64;
  const int n0 = blockIdx.x * 64;
  const int t = threadIdx.x, lane = t & 31, wave = t >> 5;
  const int mw = (wave & 1) * 32, nw = (wave >> 1) * 32;
  const float* xb = x + (size_t)b * 512 * 4096;

#if HAVE_TDM
  if (wave == 0) {
    // 64 rows x 512 elems, row pitch 520 halfwords: pad 4 DW per 256 DW.
    tdm_load_2d(lds_off(sA), wq + (size_t)m0 * 512,
                /*tensor_d0=*/512, /*tensor_d1=*/1536,
                /*tile_d0=*/512, /*tile_d1=*/64,
                /*stride0=*/512, /*pad_interval=*/7, /*pad_amount=*/3, 1);
    __builtin_amdgcn_s_wait_tensorcnt(0);
  }
#else
  { // manual full A stage: thread t copies half a row (256 halfwords)
    int r = t >> 1, h = (t & 1) * 256;
    const u32x4* src = (const u32x4*)(wq + (size_t)(m0 + r) * 512 + h);
    u32x4* dst = (u32x4*)(sA + r * 520 + h);
    #pragma unroll
    for (int i = 0; i < 32; ++i) dst[i] = src[i];
  }
#endif

  v8f acc[2][2] = {};
  for (int k0 = 0; k0 < 512; k0 += 32) {
    __syncthreads();
    { // stage B (f32 x -> bf16, transpose into [n][k])
      int c = t >> 2, seg = (t & 3) * 16;
      const f32x4* s4 = (const f32x4*)(xb + (size_t)(k0 + c) * 4096 + n0 + seg);
      #pragma unroll
      for (int i = 0; i < 4; ++i) {
        f32x4 f = s4[i];
        int n = seg + i * 4;
        sB[(n + 0) * 40 + c] = f2bf(f[0]);
        sB[(n + 1) * 40 + c] = f2bf(f[1]);
        sB[(n + 2) * 40 + c] = f2bf(f[2]);
        sB[(n + 3) * 40 + c] = f2bf(f[3]);
      }
    }
    __syncthreads();
    v16bf a0 = load_frag_a(sA + k0, 520, mw + 0,  lane);
    v16bf a1 = load_frag_a(sA + k0, 520, mw + 16, lane);
    v16bf b0 = load_frag_b(sB, 40, nw + 0,  lane);
    v16bf b1 = load_frag_b(sB, 40, nw + 16, lane);
    acc[0][0] = WMMA_BF16(a0, b0, acc[0][0]);
    acc[0][1] = WMMA_BF16(a0, b1, acc[0][1]);
    acc[1][0] = WMMA_BF16(a1, b0, acc[1][0]);
    acc[1][1] = WMMA_BF16(a1, b1, acc[1][1]);
  }
  float* outp = qkv + (size_t)b * 1536 * 4096;
  const int colL = lane & 15;
  const int rh   = (lane & 16) ? 8 : 0;
  #pragma unroll
  for (int mi = 0; mi < 2; ++mi)
    #pragma unroll
    for (int ni = 0; ni < 2; ++ni) {
      int n = n0 + nw + ni * 16 + colL;
      #pragma unroll
      for (int r = 0; r < 8; ++r) {
        int m = m0 + mw + mi * 16 + r + rh;
        outp[(size_t)m * 4096 + n] = acc[mi][ni][r];
      }
    }
}

// ---------------------------------------------------------------------------
// Kernel 3: RMS over the full 3C=1536 channel dim per (b,n); eps AFTER sqrt.
// ---------------------------------------------------------------------------
__global__ __launch_bounds__(256, 1)
void adm_rms(const float* __restrict__ qkv, unsigned short* __restrict__ qkvn) {
  int idx = blockIdx.x * 256 + threadIdx.x;   // 0 .. 16383
  int b = idx >> 12;
  int n = idx & 4095;
  const float* base = qkv + (size_t)b * 1536 * 4096 + n;
  float s = 0.f;
  for (int o = 0; o < 1536; ++o) { float v = base[(size_t)o * 4096]; s += v * v; }
  float inv = 1.0f / (sqrtf(s * (1.0f / 1536.0f)) + 1e-4f);
  unsigned short* ob = qkvn + (size_t)b * 1536 * 4096 + n;
  for (int o = 0; o < 1536; ++o) ob[(size_t)o * 4096] = f2bf(base[(size_t)o * 4096] * inv);
}

// ---------------------------------------------------------------------------
// Kernel 4: flash attention per 64-query tile. 16 waves / 512 threads.
//   S = Q^T K * C^-0.5, online softmax (P bf16 in-place over S), O += P V.
// V tile (512x64, stride 4096) staged via TDM into [c][j] pitch-72 LDS.
// Next j-tile K/V rows prefetched with global_prefetch.
// LDS: sQ 66.5KB + sBuf(K/V time-shared) 73.7KB + sSP 18.4KB + stats = 159.5KB
// ---------------------------------------------------------------------------
__global__ __launch_bounds__(512, 1)
void adm_attn(const unsigned short* __restrict__ qkvn,
              unsigned short* __restrict__ hT) {
  __shared__ unsigned short sQ[64 * 520];    // [i][c]
  __shared__ unsigned short sBuf[512 * 72];  // K phase: [j][c] ld 520 ; V phase: [c][j] ld 72
  __shared__ float sSP[64 * 72];             // S (f32) then P (bf16 in-place, ld 144 halfwords)
  __shared__ float sM[64];
  __shared__ float sL[64];
  __shared__ float sCorr[64];

  const int b  = blockIdx.y;
  const int i0 = blockIdx.x * 64;
  const int t = threadIdx.x, lane = t & 31, wave = t >> 5;
  const int is_ = (wave & 3) * 16;        // i sub-tile (shared by S and O)
  const int js_ = (wave >> 2) * 16;       // j sub-tile for S
  const int co_ = (wave >> 2) * 128;      // c chunk for O
  const unsigned short* qp = qkvn + (size_t)b * 1536 * 4096;
  const unsigned short* kp = qp + (size_t)512 * 4096;
  const unsigned short* vp = qp + (size_t)1024 * 4096;
  unsigned short* sPu = (unsigned short*)sSP;
  const float scale = 0.04419417382415922f;   // 512^-0.5

  { // stage Q tile once (persistent): sQ[i][c] = q[c][i0+i]
    const unsigned short* src = qp + (size_t)t * 4096 + i0;
    #pragma unroll
    for (int i = 0; i < 64; ++i) sQ[i * 520 + t] = src[i];
  }
  if (t < 64) { sM[t] = -3.0e38f; sL[t] = 0.0f; }
  v8f accO[8] = {};

  for (int j0 = 0; j0 < 4096; j0 += 64) {
    __syncthreads();                         // prev PV done reading sBuf(V)
    { // stage K tile: sBuf[j][c]; prefetch next tile's K/V rows
      const unsigned short* src = kp + (size_t)t * 4096 + j0;
      #pragma unroll
      for (int j = 0; j < 64; ++j) sBuf[j * 520 + t] = src[j];
      if (j0 + 64 < 4096) {
        __builtin_prefetch((const void*)(kp + (size_t)t * 4096 + j0 + 64), 0, 0);
        __builtin_prefetch((const void*)(vp + (size_t)t * 4096 + j0 + 64), 0, 0);
      }
    }
    __syncthreads();
    // ---- S = Q^T K : one 16x16 WMMA tile per wave, 16 K-steps ----
    v8f accS = {};
    #pragma unroll
    for (int k0 = 0; k0 < 512; k0 += 32) {
      v16bf a  = load_frag_a(sQ + k0,   520, is_, lane);
      v16bf bq = load_frag_b(sBuf + k0, 520, js_, lane);
      accS = WMMA_BF16(a, bq, accS);
    }
    { // spill scaled S to LDS (f32)
      int col = js_ + (lane & 15);
      int rh  = (lane & 16) ? 8 : 0;
      #pragma unroll
      for (int r = 0; r < 8; ++r) sSP[(is_ + r + rh) * 72 + col] = accS[r] * scale;
    }
    __syncthreads();                         // K consumed; S visible
#if HAVE_TDM
    if (wave == 0) {
      // V tile: 512 rows x 64 elems, row pitch 72 halfwords (pad 4 DW / 32 DW)
      tdm_load_2d(lds_off(sBuf), vp + j0,
                  /*tensor_d0=*/4096, /*tensor_d1=*/512,
                  /*tile_d0=*/64, /*tile_d1=*/512,
                  /*stride0=*/4096, /*pad_interval=*/4, /*pad_amount=*/3, 1);
      __builtin_amdgcn_s_wait_tensorcnt(0);
    }
#else
    { // restage V tile into sBuf as [c][j] (all 512 threads)
      const u32x4* src = (const u32x4*)(vp + (size_t)t * 4096 + j0);
      u32x4* dst = (u32x4*)(sBuf + t * 72);
      #pragma unroll
      for (int q8 = 0; q8 < 8; ++q8) dst[q8] = src[q8];
    }
#endif
    if (t < 64) { // online softmax, one row per thread; P -> bf16 in-place
      float* rowp = sSP + t * 72;
      float m_old = sM[t];
      float mx = m_old;
      #pragma unroll
      for (int j = 0; j < 64; ++j) mx = fmaxf(mx, rowp[j]);
      float corr = __expf(m_old - mx);
      float l = sL[t] * corr;
      unsigned short* prow = sPu + t * 144;
      #pragma unroll
      for (int j = 0; j < 64; ++j) {         // ushort write j only clobbers f32 j/2 (<j, consumed)
        float e = __expf(rowp[j] - mx);
        l += e;
        prow[j] = f2bf(e);
      }
      sM[t] = mx; sL[t] = l; sCorr[t] = corr;
    }
    __syncthreads();
    { // rescale O by correction, then O += P V (2 K-steps x 8 c-subtiles)
      int rh = (lane & 16) ? 8 : 0;
      float corr[8];
      #pragma unroll
      for (int r = 0; r < 8; ++r) corr[r] = sCorr[is_ + r + rh];
      #pragma unroll
      for (int u = 0; u < 8; ++u)
        #pragma unroll
        for (int r = 0; r < 8; ++r) accO[u][r] *= corr[r];
      #pragma unroll
      for (int k0 = 0; k0 < 64; k0 += 32) {
        v16bf a = load_frag_a(sPu + k0, 144, is_, lane);
        #pragma unroll
        for (int u = 0; u < 8; ++u) {
          v16bf bv = load_frag_b(sBuf + k0, 72, co_ + u * 16, lane);
          accO[u] = WMMA_BF16(a, bv, accO[u]);
        }
      }
    }
  }
  __syncthreads();
  { // O /= l, write hT[b][i][c] bf16 (transposed layout for out-proj B tiles)
    int rh = (lane & 16) ? 8 : 0;
    float linv[8];
    #pragma unroll
    for (int r = 0; r < 8; ++r) linv[r] = 1.0f / sL[is_ + r + rh];
    unsigned short* ob = hT + ((size_t)b * 4096 + i0) * 512;
    #pragma unroll
    for (int u = 0; u < 8; ++u) {
      int c = co_ + u * 16 + (lane & 15);
      #pragma unroll
      for (int r = 0; r < 8; ++r)
        ob[(size_t)(is_ + r + rh) * 512 + c] = f2bf(accO[u][r] * linv[r]);
    }
  }
}

// ---------------------------------------------------------------------------
// Kernel 5: out = (x + W_out @ h) * 2^-0.5, h given as hT[b][n][c] bf16.
// Both 64x512 bf16 tiles staged ONCE via TDM (2 tensor ops per block),
// then the whole K loop is pure LDS frag loads + WMMA.
// ---------------------------------------------------------------------------
__global__ __launch_bounds__(256, 1)
void adm_outproj(const unsigned short* __restrict__ wo,
                 const unsigned short* __restrict__ hT,
                 const float* __restrict__ x,
                 float* __restrict__ out) {
  __shared__ unsigned short sA[64 * 520];  // [o][c] full K extent
  __shared__ unsigned short sB[64 * 520];  // [n][c] full K extent
  const int b  = blockIdx.z;
  const int m0 = blockIdx.y * 64;
  const int n0 = blockIdx.x * 64;
  const int t = threadIdx.x, lane = t & 31, wave = t >> 5;
  const int mw = (wave & 3) * 16;
  const int nw = (wave >> 2) * 32;
  const unsigned short* hb = hT + (size_t)b * 4096 * 512;

#if HAVE_TDM
  if (wave == 0) {
    tdm_load_2d(lds_off(sA), wo + (size_t)m0 * 512,
                512, 512, 512, 64, 512, 7, 3, 1);
    tdm_load_2d(lds_off(sB), hb + (size_t)n0 * 512,
                512, 4096, 512, 64, 512, 7, 3, 1);
    __builtin_amdgcn_s_wait_tensorcnt(0);
  }
#else
  { // manual: thread t copies a quarter row (128 halfwords) of each tile
    int r = t >> 2, h = (t & 3) * 128;
    const u32x4* sa = (const u32x4*)(wo + (size_t)(m0 + r) * 512 + h);
    const u32x4* sb = (const u32x4*)(hb + (size_t)(n0 + r) * 512 + h);
    u32x4* da = (u32x4*)(sA + r * 520 + h);
    u32x4* db = (u32x4*)(sB + r * 520 + h);
    #pragma unroll
    for (int i = 0; i < 16; ++i) { da[i] = sa[i]; db[i] = sb[i]; }
  }
#endif
  __syncthreads();

  v8f acc[2] = {};
  #pragma unroll
  for (int k0 = 0; k0 < 512; k0 += 32) {
    v16bf a  = load_frag_a(sA + k0, 520, mw, lane);
    v16bf b0 = load_frag_b(sB + k0, 520, nw + 0,  lane);
    v16bf b1 = load_frag_b(sB + k0, 520, nw + 16, lane);
    acc[0] = WMMA_BF16(a, b0, acc[0]);
    acc[1] = WMMA_BF16(a, b1, acc[1]);
  }
  const float* xb = x + (size_t)b * 512 * 4096;
  float* ob = out + (size_t)b * 512 * 4096;
  const float rs2 = 0.70710678118654752f;
  const int colL = lane & 15, rh = (lane & 16) ? 8 : 0;
  #pragma unroll
  for (int ni = 0; ni < 2; ++ni) {
    int n = n0 + nw + ni * 16 + colL;
    #pragma unroll
    for (int r = 0; r < 8; ++r) {
      int m = m0 + mw + r + rh;
      size_t off = (size_t)m * 4096 + n;
      ob[off] = (xb[off] + acc[ni][r]) * rs2;
    }
  }
}

// ---------------------------------------------------------------------------
// Launch. Workspace layout (bytes):
//   [0)            w_qkv bf16  1536*512*2 = 1,572,864
//   [1,572,864)    w_out bf16   512*512*2 =   524,288
//   [2,097,152)    qkv f32   4*1536*4096*4 = 100,663,296
//   [102,760,448)  qkvn bf16 4*1536*4096*2 =  50,331,648
//   [153,092,096)  hT bf16   4*4096*512*2  =  16,777,216   (total ~170 MB)
// ---------------------------------------------------------------------------
extern "C" void kernel_launch(void* const* d_in, const int* in_sizes, int n_in,
                              void* d_out, int out_size, void* d_ws, size_t ws_size,
                              hipStream_t stream) {
  const float* x     = (const float*)d_in[0];
  const float* v_qkv = (const float*)d_in[1];
  const float* g_qkv = (const float*)d_in[2];
  const float* v_out = (const float*)d_in[3];
  const float* g_out = (const float*)d_in[4];
  float* out = (float*)d_out;
  char* ws = (char*)d_ws;
  unsigned short* wq   = (unsigned short*)(ws + (size_t)0);
  unsigned short* wo   = (unsigned short*)(ws + (size_t)1572864);
  float*          qkv  = (float*)         (ws + (size_t)2097152);
  unsigned short* qkvn = (unsigned short*)(ws + (size_t)102760448);
  unsigned short* hT   = (unsigned short*)(ws + (size_t)153092096);

  adm_wnorm<<<1536, 128, 0, stream>>>(v_qkv, g_qkv, wq, 512);
  adm_wnorm<<<512,  128, 0, stream>>>(v_out, g_out, wo, 512);
  adm_qkv_gemm<<<dim3(64, 24, 4), 128, 0, stream>>>(wq, x, qkv);
  adm_rms<<<64, 256, 0, stream>>>(qkv, qkvn);
  adm_attn<<<dim3(64, 4), 512, 0, stream>>>(qkvn, hT);
  adm_outproj<<<dim3(64, 8, 4), 256, 0, stream>>>(wo, hT, x, out);

  (void)in_sizes; (void)n_in; (void)out_size; (void)ws_size;
}
